// AutoEncoderRNN_27049704030727
// MI455X (gfx1250) — matrix operand order repaired and checked by reading, not compiled, not run on hardware
//
#include <hip/hip_runtime.h>

#define NWG   64
#define BLOCK 256
#define Bsz   256
#define Tsz   512
#define Hsz   1024

typedef __attribute__((ext_vector_type(16))) __bf16   v16bf;
typedef __attribute__((ext_vector_type(8)))  float    v8f;
typedef __attribute__((ext_vector_type(4)))  unsigned u32x4;

union V16 { v16bf v; u32x4 q[2]; };
static_assert(sizeof(V16) == 32, "V16 must be 32 bytes");

__device__ __forceinline__ unsigned short f2bf(float f) {
    unsigned u = __float_as_uint(f);
    u += 0x7FFFu + ((u >> 16) & 1u);   // round-to-nearest-even
    return (unsigned short)(u >> 16);
}
__device__ __forceinline__ float bf2f(unsigned short s) {
    return __uint_as_float(((unsigned)s) << 16);
}

// Branch-free tanh: 1 - 2/(exp(2x)+1). Saturates to +/-1, no NaN paths.
// v_exp_f32 + v_rcp_f32 instead of libm's exec-mask-divergent slow path.
__device__ __forceinline__ float fast_tanh(float x) {
    float e = __expf(2.0f * x);
    return 1.0f - 2.0f * __builtin_amdgcn_rcpf(e + 1.0f);
}
__device__ __forceinline__ float fast_sigmoid(float x) {
    return __builtin_amdgcn_rcpf(1.0f + __expf(-x));
}

// Monotonic-counter grid barrier. Counter is zeroed by hipMemsetAsync before
// each kernel launch (graph-capturable), so target arithmetic is exact.
__device__ __forceinline__ void grid_barrier(unsigned* cnt) {
    __syncthreads();
    if (threadIdx.x == 0) {
        __threadfence();                                   // release my writes
        unsigned prev   = atomicAdd(cnt, 1u);
        unsigned target = (prev / NWG + 1u) * NWG;
        while (__hip_atomic_load(cnt, __ATOMIC_RELAXED,
                                 __HIP_MEMORY_SCOPE_AGENT) < target) {
            __builtin_amdgcn_s_sleep(2);
        }
        __threadfence();                                   // acquire others'
    }
    __syncthreads();
}

// One RNN matmul step for this workgroup's 16 output columns.
//   hnxt[b, colbase+jl] = tanh( (h @ Whh^T)[b,j] + x[b]*Wih[j] + bsum[j] )
// and (DEC) accumulates  sum_j hnxt[b,j]*Wout[j]  into lds_xpart[b].
template <bool DEC>
__device__ __forceinline__ void rnn_step(
    const unsigned short* __restrict__ hcur,   // global bf16 (B x H)
    unsigned short*       __restrict__ hnxt,   // global bf16 (B x H)
    const unsigned short* __restrict__ lds_w,  // 16 x H bf16 Whh slice
    const float* __restrict__ lds_wih,
    const float* __restrict__ lds_bsum,
    const float* __restrict__ lds_wout,
    const float* __restrict__ lds_x,
    float*       __restrict__ lds_xpart,
    int colbase)
{
    const int lane  = threadIdx.x & 31;
    const int wave  = threadIdx.x >> 5;
    const int mbase = wave * 32;              // 8 waves x 32 rows = 256 rows
    const int jl    = lane & 15;              // output column within slice / A row
    const int hi    = lane >> 4;              // half-wave select

    // A (16x32 bf16) per-lane layout: lane<16 -> K {0..7,16..23}, lane>=16 -> K {8..15,24..31}
    const int k0a = hi * 8;
    // B (32x16 bf16) per-lane layout: lane<16 -> K {0..7,16..23}, lane>=16 -> K {8..15,24..31}
    const int kb  = hi * 8;

    v8f acc0 = {}; v8f acc1 = {};
    const unsigned short* a0 = hcur + (size_t)(mbase + jl) * Hsz + k0a;
    const unsigned short* a1 = a0 + 16 * Hsz;
    const unsigned short* wr = lds_w + jl * Hsz + kb;

#pragma unroll 4
    for (int ks = 0; ks < Hsz / 32; ++ks) {
        V16 Bv, A0, A1;
        Bv.q[0] = *(const u32x4*)(wr);
        Bv.q[1] = *(const u32x4*)(wr + 16);       // K + 16
        A0.q[0] = *(const u32x4*)(a0);
        A0.q[1] = *(const u32x4*)(a0 + 16);       // K + 16
        A1.q[0] = *(const u32x4*)(a1);
        A1.q[1] = *(const u32x4*)(a1 + 16);
        acc0 = __builtin_amdgcn_wmma_f32_16x16x32_bf16(false, A0.v, false, Bv.v,
                                                       (short)0, acc0, false, false);
        acc1 = __builtin_amdgcn_wmma_f32_16x16x32_bf16(false, A1.v, false, Bv.v,
                                                       (short)0, acc1, false, false);
        a0 += 32; a1 += 32; wr += 32;
    }

    // D layout: VGPR r -> M = r (lanes 0-15) or M = 8+r (lanes 16-31); N = lane&15
    const int roff = hi * 8;
#pragma unroll
    for (int mt = 0; mt < 2; ++mt) {
        v8f acc = mt ? acc1 : acc0;
#pragma unroll
        for (int r = 0; r < 8; ++r) {
            int   row = mbase + mt * 16 + roff + r;
            float pre = acc[r] + lds_x[row] * lds_wih[jl] + lds_bsum[jl];
            float hv  = fast_tanh(pre);
            hnxt[(size_t)row * Hsz + colbase + jl] = f2bf(hv);
            if (DEC) atomicAdd(&lds_xpart[row], hv * lds_wout[jl]);
        }
    }
}

__global__ __launch_bounds__(BLOCK) void autoenc_rnn_kernel(
    const float* __restrict__ input, const int* __restrict__ seq_len,
    const float* __restrict__ Wih_e, const float* __restrict__ Whh_e,
    const float* __restrict__ bih_e, const float* __restrict__ bhh_e,
    const float* __restrict__ Wenc,  const float* __restrict__ benc,
    const float* __restrict__ Wdec,  const float* __restrict__ bdec,
    const float* __restrict__ Wih_d, const float* __restrict__ Whh_d,
    const float* __restrict__ bih_d, const float* __restrict__ bhh_d,
    const float* __restrict__ Wout,  const float* __restrict__ bout,
    float* __restrict__ out,
    unsigned* bar,                 // grid-barrier counter (zeroed per launch)
    float* xb,                     // 3 x 256 rotating x accumulators (zeroed)
    float* feat,                   // 256 x 3 features
    unsigned short* hb0,           // bf16 hidden state ping (zeroed: h0_enc)
    unsigned short* hb1)           // bf16 hidden state pong
{
    __shared__ __attribute__((aligned(16))) unsigned short lds_w[16 * Hsz]; // 32 KB
    __shared__ float lds_wih[16], lds_bsum[16], lds_wout[16];
    __shared__ float lds_x[Bsz];      // per-step x input (also reused for loss reduce)
    __shared__ float lds_xpart[Bsz];  // decoder per-WG partial of h@Wout^T

    const int wg      = blockIdx.x;
    const int tid     = threadIdx.x;
    const int colbase = wg * 16;

    // copy input -> out[1..] (output tuple: loss, input, output, features)
    for (int i = wg * BLOCK + tid; i < Bsz * Tsz; i += NWG * BLOCK)
        out[1 + i] = input[i];

    // ---- preload encoder weight slice (f32 -> bf16 in LDS) ----
    for (int i = tid; i < 16 * Hsz; i += BLOCK) {
        int jlo = i >> 10, k = i & (Hsz - 1);
        lds_w[i] = f2bf(Whh_e[(size_t)(colbase + jlo) * Hsz + k]);
    }
    if (tid < 16) {
        int j = colbase + tid;
        lds_wih[tid]  = Wih_e[j];
        lds_bsum[tid] = bih_e[j] + bhh_e[j];
        lds_wout[tid] = Wout[j];
    }
    __syncthreads();

    // ---------------- encoder: T sequential steps ----------------
    for (int t = 0; t < Tsz; ++t) {
        const unsigned short* hc = (t & 1) ? hb1 : hb0;   // ternary keeps global
        unsigned short*       hn = (t & 1) ? hb0 : hb1;   // addrspace inference
        lds_x[tid] = input[(size_t)tid * Tsz + t];        // x_t per batch row
        __syncthreads();
        rnn_step<false>(hc, hn, lds_w, lds_wih, lds_bsum, lds_wout,
                        lds_x, lds_xpart, colbase);
        grid_barrier(bar);
    }
    const unsigned short* hT = hb0;                       // T even -> final h in hb0

    // ---- features = sigmoid(hT @ Wenc^T + benc)  (WG0) ----
    if (wg == 0) {
        int b = tid;
        float f0 = 0.f, f1 = 0.f, f2 = 0.f;
        for (int k = 0; k < Hsz; ++k) {
            float hv = bf2f(hT[(size_t)b * Hsz + k]);
            f0 += hv * Wenc[k];
            f1 += hv * Wenc[Hsz + k];
            f2 += hv * Wenc[2 * Hsz + k];
        }
        f0 = fast_sigmoid(f0 + benc[0]);
        f1 = fast_sigmoid(f1 + benc[1]);
        f2 = fast_sigmoid(f2 + benc[2]);
        feat[b * 3 + 0] = f0; feat[b * 3 + 1] = f1; feat[b * 3 + 2] = f2;
        out[1 + 2 * Bsz * Tsz + b * 3 + 0] = f0;
        out[1 + 2 * Bsz * Tsz + b * 3 + 1] = f1;
        out[1 + 2 * Bsz * Tsz + b * 3 + 2] = f2;
    }

    // ---- swap LDS weights to decoder slice (overlaps WG0's feature work) ----
    for (int i = tid; i < 16 * Hsz; i += BLOCK) {
        int jlo = i >> 10, k = i & (Hsz - 1);
        lds_w[i] = f2bf(Whh_d[(size_t)(colbase + jlo) * Hsz + k]);
    }
    if (tid < 16) {
        int j = colbase + tid;
        lds_wih[tid]  = Wih_d[j];
        lds_bsum[tid] = bih_d[j] + bhh_d[j];
    }
    grid_barrier(bar);   // features visible; LDS swap done (includes __syncthreads)

    // ---- h0_dec = features @ Wdec^T + bdec -> hb1 (decoder t=0 reads hb1) ----
    {
        int b = tid;
        float f0 = feat[b * 3], f1 = feat[b * 3 + 1], f2 = feat[b * 3 + 2];
        for (int jlo = 0; jlo < 16; ++jlo) {
            int j = colbase + jlo;
            float h0 = bdec[j] + f0 * Wdec[j * 3] + f1 * Wdec[j * 3 + 1]
                                + f2 * Wdec[j * 3 + 2];
            hb1[(size_t)b * Hsz + j] = f2bf(h0);
        }
    }
    grid_barrier(bar);

    // ---------------- decoder: T sequential steps with x feedback ----------------
    const float bo = bout[0];
    for (int t = 0; t < Tsz; ++t) {
        float* xc = xb + (t % 3) * Bsz;            // x input accumulator (sealed)
        float* xn = xb + ((t + 1) % 3) * Bsz;      // this step's output accumulator
        float* xz = xb + ((t + 2) % 3) * Bsz;      // free: zero for step t+1

        if (wg == 0 && t > 0)                      // emit output column t-1
            out[1 + Bsz * Tsz + (size_t)tid * Tsz + (t - 1)] = xc[tid] + bo;
        lds_x[tid]     = (t == 0) ? 0.f : (xc[tid] + bo);
        lds_xpart[tid] = 0.f;
        __syncthreads();

        const unsigned short* hc = (t & 1) ? hb0 : hb1;   // read h_{t-1}
        unsigned short*       hn = (t & 1) ? hb1 : hb0;   // write h_t
        rnn_step<true>(hc, hn, lds_w, lds_wih, lds_bsum, lds_wout,
                       lds_x, lds_xpart, colbase);
        __syncthreads();

        atomicAdd(&xn[tid], lds_xpart[tid]);       // one global f32 atomic / row / WG
        if (wg == 1) xz[tid] = 0.f;                // recycle third buffer
        grid_barrier(bar);
    }

    // ---- final output column + masked MSE loss (WG0) ----
    if (wg == 0) {
        int b = tid;
        out[1 + Bsz * Tsz + (size_t)b * Tsz + (Tsz - 1)] = xb[2 * Bsz + b] + bo;

        int len = seq_len[b];
        const float* ib = input + (size_t)b * Tsz;
        const float* ob = out + 1 + Bsz * Tsz + (size_t)b * Tsz;
        float s = 0.f;
        for (int t2 = 0; t2 < len; ++t2) { float d = ib[t2] - ob[t2]; s += d * d; }
        lds_x[b]     = s;            // reuse LDS for reduction
        lds_xpart[b] = (float)len;
        __syncthreads();
        if (tid == 0) {
            float ss = 0.f, cs = 0.f;
            for (int i = 0; i < Bsz; ++i) { ss += lds_x[i]; cs += lds_xpart[i]; }
            out[0] = ss / cs;
        }
    }
}

extern "C" void kernel_launch(void* const* d_in, const int* in_sizes, int n_in,
                              void* d_out, int out_size, void* d_ws, size_t ws_size,
                              hipStream_t stream) {
    (void)in_sizes; (void)n_in; (void)out_size; (void)ws_size;
    unsigned char* ws = (unsigned char*)d_ws;
    unsigned*       bar  = (unsigned*)ws;
    float*          xb   = (float*)(ws + 256);    // 3 x 256 f32
    float*          feat = (float*)(ws + 4096);   // 256 x 3 f32
    unsigned short* hb0  = (unsigned short*)(ws + 8192);
    unsigned short* hb1  = hb0 + (size_t)Bsz * Hsz;

    // Zero ctrl block (barrier counter, x accumulators) + both h buffers
    // (h0_enc must be zero). hipMemsetAsync is graph-capture legal.
    size_t zero_bytes = 8192 + 2 * (size_t)Bsz * Hsz * sizeof(unsigned short);
    hipMemsetAsync(d_ws, 0, zero_bytes, stream);

    autoenc_rnn_kernel<<<NWG, BLOCK, 0, stream>>>(
        (const float*)d_in[0],  (const int*)d_in[1],
        (const float*)d_in[2],  (const float*)d_in[3],
        (const float*)d_in[4],  (const float*)d_in[5],
        (const float*)d_in[6],  (const float*)d_in[7],
        (const float*)d_in[8],  (const float*)d_in[9],
        (const float*)d_in[10], (const float*)d_in[11],
        (const float*)d_in[12], (const float*)d_in[13],
        (const float*)d_in[14], (const float*)d_in[15],
        (float*)d_out, bar, xb, feat, hb0, hb1);
}
